// MiniMaxM2DecoderLayer_88579405513105
// MI455X (gfx1250) — compile-verified, matrix operand order checked by reading, not run on hardware
//
#include <hip/hip_runtime.h>
#include <hip/hip_bf16.h>

// ---------------------------------------------------------------------------
// MiniMax-M2 decoder layer for gfx1250 (MI455X), bf16 WMMA everywhere,
// TDM (tensor_load_to_lds) staged GEMMs, flash sliding-window attention.
// B=2, S=2048, HID=2048, H=16, HK=4, D=128, RD=64, IM=8192, WIN=512
// ---------------------------------------------------------------------------

#define B_  2
#define S_  2048
#define HID_ 2048
#define H_  16
#define HK_ 4
#define D_  128
#define RD_ 64
#define IM_ 8192
#define WIN_ 512
#define EPS_ 1e-6f
#define SCALE_ 0.08838834764831845f   // 128^-0.5
#define NTOK (B_*S_)                  // 4096

typedef __attribute__((ext_vector_type(16))) __bf16       v16bf;
typedef __attribute__((ext_vector_type(8)))  float        v8f;
typedef __attribute__((ext_vector_type(4)))  unsigned int v4u;
typedef __attribute__((ext_vector_type(4)))  unsigned int u32x4;
typedef __attribute__((ext_vector_type(8)))  int          i32x8;
typedef __attribute__((ext_vector_type(4)))  int          i32x4;

union Frag { v16bf b; v4u u[2]; };

#if defined(__has_builtin)
#if __has_builtin(__builtin_amdgcn_tensor_load_to_lds) && \
    __has_builtin(__builtin_amdgcn_s_wait_tensorcnt)
#define USE_TDM 1
#endif
#endif
#ifndef USE_TDM
#define USE_TDM 0
#endif

__device__ __forceinline__ unsigned short f2bf(float x) {
    unsigned int u = __builtin_bit_cast(unsigned int, x);
    u += 0x7fffu + ((u >> 16) & 1u);       // round-to-nearest-even
    return (unsigned short)(u >> 16);
}
__device__ __forceinline__ float bf2f(unsigned short u) {
    return __builtin_bit_cast(float, (unsigned int)u << 16);
}

__device__ __forceinline__ v8f wmma_bf16(v16bf a, v16bf b, v8f c) {
    return __builtin_amdgcn_wmma_f32_16x16x32_bf16(
        /*neg_a=*/false, a, /*neg_b=*/false, b,
        /*c_mod=*/(short)0, c, /*reuse_a=*/false, /*reuse_b=*/false);
}

#if USE_TDM
// ---------------------------------------------------------------------------
// TDM: DMA a [tile_d1=128 rows x tile_d0=32 elems] bf16 tile (row stride
// `stride_elems` in memory) into LDS at byte offset lds_off, inserting 16B of
// LDS padding after every 64B row => LDS row pitch 80B (matches lA[][40]).
// pad_interval field 3 -> 16 DWORDs (64B); pad_amount field 3 -> 4 DWORDs.
// Issued by one wave; tracked on TENSORcnt.
// ---------------------------------------------------------------------------
__device__ __forceinline__ void tdm_load_tile_128x32(
    unsigned lds_off, const unsigned short* g, long stride_elems) {
    unsigned long long ga = (unsigned long long)(uintptr_t)g;
    u32x4 g0;
    g0[0] = 1u;                                     // count=1 (valid D#)
    g0[1] = lds_off;                                // lds_addr
    g0[2] = (unsigned)ga;                           // global_addr[31:0]
    g0[3] = (unsigned)(ga >> 32) | (2u << 30);      // global_addr[56:32] | type=2
    const unsigned T0 = 0x40000000u;                // huge tensor dims (no clip)
    const unsigned T1 = 0x40000000u;
    i32x8 g1;
    g1[0] = (int)((1u << 16) | (1u << 20) | (3u << 22) | (3u << 25));
            // data_size=2B | pad_enable | pad_interval=16dw | pad_amount=4dw
    g1[1] = (int)((T0 & 0xFFFFu) << 16);            // tensor_dim0 lo16
    g1[2] = (int)((T0 >> 16) | ((T1 & 0xFFFFu) << 16));
    g1[3] = (int)((T1 >> 16) | (32u << 16));        // tile_dim0 = 32
    g1[4] = 128;                                    // tile_dim1 = 128
    g1[5] = (int)(unsigned)stride_elems;            // tensor_dim0_stride lo32
    g1[6] = (int)(unsigned)(((unsigned long long)stride_elems >> 32) & 0xFFFFu);
    g1[7] = 0;
    const i32x4 z4 = {0, 0, 0, 0};
#if __clang_major__ >= 23
    const i32x8 z8 = {0, 0, 0, 0, 0, 0, 0, 0};
    __builtin_amdgcn_tensor_load_to_lds(g0, g1, z4, z4, z8, 0);
#else
    __builtin_amdgcn_tensor_load_to_lds(g0, g1, z4, z4, 0);
#endif
}
#endif

// ---------------------------------------------------------------------------
// f32 -> bf16 cast (weights)
// ---------------------------------------------------------------------------
__global__ __launch_bounds__(256) void cast_f32_bf16(
    const float* __restrict__ in, unsigned short* __restrict__ out, long n) {
    long stride = (long)gridDim.x * 256;
    for (long i = (long)blockIdx.x * 256 + threadIdx.x; i < n; i += stride)
        out[i] = f2bf(in[i]);
}

// ---------------------------------------------------------------------------
// RMSNorm(f32 row) -> bf16 row.  One block per row.
// ---------------------------------------------------------------------------
__global__ __launch_bounds__(256) void rmsnorm_bf16(
    const float* __restrict__ x, const float* __restrict__ w,
    unsigned short* __restrict__ out, int C) {
    __shared__ float red[8];
    const int row = blockIdx.x;
    const float* xr = x + (size_t)row * C;
    float ss = 0.f;
    for (int c = threadIdx.x; c < C; c += 256) { float v = xr[c]; ss += v * v; }
    #pragma unroll
    for (int m = 16; m >= 1; m >>= 1) ss += __shfl_xor(ss, m, 32);
    if ((threadIdx.x & 31) == 0) red[threadIdx.x >> 5] = ss;
    __syncthreads();
    float tot = 0.f;
    #pragma unroll
    for (int i = 0; i < 8; i++) tot += red[i];
    const float rs = rsqrtf(tot / (float)C + EPS_);
    unsigned short* orow = out + (size_t)row * C;
    for (int c = threadIdx.x; c < C; c += 256)
        orow[c] = f2bf(xr[c] * rs * w[c]);
}

// ---------------------------------------------------------------------------
// RMSNorm over full row + partial RoPE (first RD dims of each head) -> bf16.
// ---------------------------------------------------------------------------
__global__ __launch_bounds__(256) void qknorm_rope_bf16(
    const float* __restrict__ x, const float* __restrict__ w,
    unsigned short* __restrict__ out, int C) {
    __shared__ float red[8];
    const int row = blockIdx.x;
    const int pos = row & (S_ - 1);
    const float* xr = x + (size_t)row * C;
    float ss = 0.f;
    for (int c = threadIdx.x; c < C; c += 256) { float v = xr[c]; ss += v * v; }
    #pragma unroll
    for (int m = 16; m >= 1; m >>= 1) ss += __shfl_xor(ss, m, 32);
    if ((threadIdx.x & 31) == 0) red[threadIdx.x >> 5] = ss;
    __syncthreads();
    float tot = 0.f;
    #pragma unroll
    for (int i = 0; i < 8; i++) tot += red[i];
    const float rs = rsqrtf(tot / (float)C + EPS_);
    unsigned short* orow = out + (size_t)row * C;
    for (int c = threadIdx.x; c < C; c += 256) {
        const int d = c & (D_ - 1);
        float xn = xr[c] * rs * w[c];
        float o;
        if (d < RD_) {
            const int j = d & 31;
            const float inv = __powf(10000.f, -(2.f * (float)j) / (float)RD_);
            const float ang = (float)pos * inv;
            const float cs = __cosf(ang), sn = __sinf(ang);
            if (d < 32) {
                float xp = xr[c + 32] * rs * w[c + 32];
                o = xn * cs - xp * sn;
            } else {
                float xp = xr[c - 32] * rs * w[c - 32];
                o = xn * cs + xp * sn;
            }
        } else {
            o = xn;
        }
        orow[c] = f2bf(o);
    }
}

// ---------------------------------------------------------------------------
// V transpose: vb [B*S, HK*D] -> vt [(b*HK+hk)*D + d][S]   (bf16)
// ---------------------------------------------------------------------------
__global__ __launch_bounds__(256) void v_transpose(
    const unsigned short* __restrict__ vb, unsigned short* __restrict__ vt) {
    const long n = (long)NTOK * (HK_ * D_);
    const long stride = (long)gridDim.x * 256;
    for (long i = (long)blockIdx.x * 256 + threadIdx.x; i < n; i += stride) {
        const int col = (int)(i & (HK_ * D_ - 1));  // hk*128 + d
        const long row = i >> 9;                    // b*S + s
        const int b = (int)(row >> 11);
        const int s = (int)(row & (S_ - 1));
        vt[((long)(b * HK_ * D_ + col)) * S_ + s] = vb[i];
    }
}

// ---------------------------------------------------------------------------
// bf16 GEMM:  C[M,N] = A[M,K] @ W[N,K]^T     (both operands K-contiguous)
// block 256 thr (8 waves), block tile 128x128, wave tile 32x64, K-step 32.
// TDM path: wave 0 DMAs both 128x32 tiles per K-step into double-buffered
// padded LDS; compute overlaps the DMA; TENSORcnt + 1 barrier per step.
// MODE 0: f32 out.  MODE 1: bf16 out.  MODE 2: f32 out + f32 residual.
// ---------------------------------------------------------------------------
template <int MODE>
__global__ __launch_bounds__(256) void gemm_bf16_tn(
    const unsigned short* __restrict__ A,
    const unsigned short* __restrict__ Bw,
    float* __restrict__ Cf, unsigned short* __restrict__ Cb,
    const float* __restrict__ R,
    int M, int N, int K) {
    const int tid  = threadIdx.x;
    const int lane = tid & 31;
    const int w    = tid >> 5;
    const int wm   = w & 3;        // 4 waves in M
    const int wn   = w >> 2;       // 2 waves in N
    const int nl   = lane & 15;
    const int hi   = lane >> 4;
    const int m0   = blockIdx.y * 128;
    const int n0   = blockIdx.x * 128;

    v8f acc[2][4];
    const v8f vzero = {};
    #pragma unroll
    for (int i = 0; i < 2; i++)
        #pragma unroll
        for (int j = 0; j < 4; j++) acc[i][j] = vzero;

#if USE_TDM
    __shared__ unsigned short lA[2][128][40];   // 80B row pitch (64B + 16B pad)
    __shared__ unsigned short lB[2][128][40];
    const unsigned ldsA0 = (unsigned)(uintptr_t)&lA[0][0][0];
    const unsigned ldsB0 = (unsigned)(uintptr_t)&lB[0][0][0];
    const unsigned bufBytes = 128u * 40u * 2u;

    if (w == 0) {
        tdm_load_tile_128x32(ldsA0, A  + (size_t)m0 * K, K);
        tdm_load_tile_128x32(ldsB0, Bw + (size_t)n0 * K, K);
        __builtin_amdgcn_s_wait_tensorcnt(0);
    }
    __syncthreads();

    int buf = 0;
    for (int k0 = 0; k0 < K; k0 += 32) {
        if (w == 0 && k0 + 32 < K) {    // prefetch next tiles via TDM
            tdm_load_tile_128x32(ldsA0 + (buf ^ 1) * bufBytes,
                                 A + (size_t)m0 * K + k0 + 32, K);
            tdm_load_tile_128x32(ldsB0 + (buf ^ 1) * bufBytes,
                                 Bw + (size_t)n0 * K + k0 + 32, K);
        }
        Frag af[2], bf[4];
        #pragma unroll
        for (int i = 0; i < 2; i++) {
            const int r = wm * 32 + i * 16 + nl;
            af[i].u[0] = *(const v4u*)&lA[buf][r][hi * 8];
            af[i].u[1] = *(const v4u*)&lA[buf][r][hi * 8 + 16];
        }
        #pragma unroll
        for (int j = 0; j < 4; j++) {
            const int r = wn * 64 + j * 16 + nl;
            bf[j].u[0] = *(const v4u*)&lB[buf][r][hi * 8];
            bf[j].u[1] = *(const v4u*)&lB[buf][r][hi * 8 + 16];
        }
        #pragma unroll
        for (int i = 0; i < 2; i++)
            #pragma unroll
            for (int j = 0; j < 4; j++)
                acc[i][j] = wmma_bf16(af[i].b, bf[j].b, acc[i][j]);

        __builtin_amdgcn_s_wait_tensorcnt(0);  // wave0: next tiles landed
        __syncthreads();
        buf ^= 1;
    }
#else
    __shared__ unsigned short lA[128][40];
    __shared__ unsigned short lB[128][40];
    const int lrow = tid >> 1;          // 0..127
    const int lcol = (tid & 1) << 4;    // 0 or 16

    for (int k0 = 0; k0 < K; k0 += 32) {
        const unsigned short* gA = A  + (size_t)(m0 + lrow) * K + k0 + lcol;
        const unsigned short* gB = Bw + (size_t)(n0 + lrow) * K + k0 + lcol;
        const v4u a0 = *(const v4u*)gA;
        const v4u a1 = *(const v4u*)(gA + 8);
        const v4u b0 = *(const v4u*)gB;
        const v4u b1 = *(const v4u*)(gB + 8);
        __syncthreads();
        *(v4u*)&lA[lrow][lcol]     = a0;
        *(v4u*)&lA[lrow][lcol + 8] = a1;
        *(v4u*)&lB[lrow][lcol]     = b0;
        *(v4u*)&lB[lrow][lcol + 8] = b1;
        __syncthreads();

        Frag af[2], bf[4];
        #pragma unroll
        for (int i = 0; i < 2; i++) {
            const int r = wm * 32 + i * 16 + nl;
            af[i].u[0] = *(const v4u*)&lA[r][hi * 8];
            af[i].u[1] = *(const v4u*)&lA[r][hi * 8 + 16];
        }
        #pragma unroll
        for (int j = 0; j < 4; j++) {
            const int r = wn * 64 + j * 16 + nl;
            bf[j].u[0] = *(const v4u*)&lB[r][hi * 8];
            bf[j].u[1] = *(const v4u*)&lB[r][hi * 8 + 16];
        }
        #pragma unroll
        for (int i = 0; i < 2; i++)
            #pragma unroll
            for (int j = 0; j < 4; j++)
                acc[i][j] = wmma_bf16(af[i].b, bf[j].b, acc[i][j]);
    }
#endif

    #pragma unroll
    for (int i = 0; i < 2; i++)
        #pragma unroll
        for (int j = 0; j < 4; j++) {
            const int gr = m0 + wm * 32 + i * 16 + 8 * hi;
            const int gc = n0 + wn * 64 + j * 16 + nl;
            #pragma unroll
            for (int e = 0; e < 8; e++) {
                const size_t idx = (size_t)(gr + e) * N + gc;
                const float v = acc[i][j][e];
                if (MODE == 0) Cf[idx] = v;
                else if (MODE == 1) Cb[idx] = f2bf(v);
                else Cf[idx] = R[idx] + v;
            }
        }
}

// ---------------------------------------------------------------------------
// Sliding-window causal flash attention (bf16 WMMA, f32 softmax).
// Block = 128 thr (4 waves); wave owns a 16-row q tile; key tiles of 32.
// ---------------------------------------------------------------------------
__global__ __launch_bounds__(128) void attn_swa(
    const unsigned short* __restrict__ Q,
    const unsigned short* __restrict__ Kb,
    const unsigned short* __restrict__ Vt,
    unsigned short* __restrict__ O) {
    __shared__ unsigned short pst[4][16][40];   // per-wave P staging (padded)

    const int tid  = threadIdx.x;
    const int lane = tid & 31;
    const int w    = tid >> 5;
    const int nl   = lane & 15;
    const int hi   = lane >> 4;

    const int bid  = blockIdx.x;
    const int qblk = bid & 31;            // S/64 = 32
    const int h    = (bid >> 5) & 15;
    const int b    = bid >> 9;
    const int q0   = qblk * 64 + w * 16;
    const int hk   = h >> 2;              // nrep = 4

    // ----- Q fragments: 16x128 split into 4 K-chunks of 32 -----
    Frag qf[4];
    const unsigned short* qbase =
        Q + (size_t)(b * S_ + q0 + nl) * (H_ * D_) + h * D_;
    #pragma unroll
    for (int c = 0; c < 4; c++) {
        const int kc = c * 32 + hi * 8;
        qf[c].u[0] = *(const v4u*)(qbase + kc);
        qf[c].u[1] = *(const v4u*)(qbase + kc + 16);
    }

    v8f oacc[8];
    const v8f vzero = {};
    #pragma unroll
    for (int nc = 0; nc < 8; nc++) oacc[nc] = vzero;
    float mr[8], lr[8];
    #pragma unroll
    for (int i = 0; i < 8; i++) { mr[i] = -1e30f; lr[i] = 0.f; }

    int kstart = q0 - WIN_;
    if (kstart < 0) kstart = 0;
    kstart &= ~31;

    for (int kb0 = kstart; kb0 <= q0 + 15; kb0 += 32) {
        // prefetch next key tile into the cache hierarchy
        if (kb0 + 32 <= q0 + 15) {
            __builtin_prefetch(
                Kb + (size_t)(b * S_ + kb0 + 32 + nl) * (HK_ * D_) + hk * D_,
                0, 3);
            __builtin_prefetch(
                Vt + ((size_t)(b * HK_ * D_ + hk * D_ + nl)) * S_ + kb0 + 32,
                0, 3);
        }
        // ----- scores: S0 = Q @ K[kb0..kb0+15]^T, S1 = next 16 keys -----
        v8f s0 = vzero, s1 = vzero;
        #pragma unroll
        for (int c = 0; c < 4; c++) {
            const int kc = c * 32 + hi * 8;
            const unsigned short* kp0 =
                Kb + (size_t)(b * S_ + kb0 + nl) * (HK_ * D_) + hk * D_ + kc;
            const unsigned short* kp1 = kp0 + (size_t)16 * (HK_ * D_);
            Frag kf0, kf1;
            kf0.u[0] = *(const v4u*)kp0;
            kf0.u[1] = *(const v4u*)(kp0 + 16);
            kf1.u[0] = *(const v4u*)kp1;
            kf1.u[1] = *(const v4u*)(kp1 + 16);
            s0 = wmma_bf16(qf[c].b, kf0.b, s0);
            s1 = wmma_bf16(qf[c].b, kf1.b, s1);
        }

        // ----- online softmax (row stats live in 16-lane halves) -----
        float alpha[8];
        #pragma unroll
        for (int i = 0; i < 8; i++) {
            const int r  = i + 8 * hi;
            const int qg = q0 + r;
            const int k0g = kb0 + nl;
            const int k1g = kb0 + 16 + nl;
            float a = s0[i] * SCALE_;
            float c = s1[i] * SCALE_;
            if (k0g > qg || k0g < qg - WIN_) a = -INFINITY;
            if (k1g > qg || k1g < qg - WIN_) c = -INFINITY;
            float t = fmaxf(a, c);
            #pragma unroll
            for (int m = 8; m >= 1; m >>= 1) t = fmaxf(t, __shfl_xor(t, m, 32));
            const float mn = fmaxf(mr[i], t);
            const float al = __expf(mr[i] - mn);
            mr[i] = mn;
            const float p0 = __expf(a - mn);
            const float p1 = __expf(c - mn);
            float rsum = p0 + p1;
            #pragma unroll
            for (int m = 8; m >= 1; m >>= 1) rsum += __shfl_xor(rsum, m, 32);
            lr[i] = lr[i] * al + rsum;
            alpha[i] = al;
            pst[w][r][nl]      = f2bf(p0);
            pst[w][r][nl + 16] = f2bf(p1);
        }
        #pragma unroll
        for (int nc = 0; nc < 8; nc++)
            #pragma unroll
            for (int i = 0; i < 8; i++) oacc[nc][i] *= alpha[i];

        asm volatile("s_wait_dscnt 0x0" ::: "memory");

        // ----- P as A-fragment (16x32) from LDS -----
        Frag pf;
        pf.u[0] = *(const v4u*)&pst[w][nl][hi * 8];
        pf.u[1] = *(const v4u*)&pst[w][nl][hi * 8 + 16];

        // ----- O += P @ V  (V^T layout makes B-frags contiguous) -----
        #pragma unroll
        for (int nc = 0; nc < 8; nc++) {
            const unsigned short* vp =
                Vt + ((size_t)(b * HK_ * D_ + hk * D_ + nc * 16 + nl)) * S_ +
                kb0 + hi * 8;
            Frag vf;
            vf.u[0] = *(const v4u*)vp;
            vf.u[1] = *(const v4u*)(vp + 16);
            oacc[nc] = wmma_bf16(pf.b, vf.b, oacc[nc]);
        }
    }

    // ----- normalize and write out -----
    #pragma unroll
    for (int nc = 0; nc < 8; nc++)
        #pragma unroll
        for (int i = 0; i < 8; i++) {
            const int r = i + 8 * hi;
            const size_t idx =
                (size_t)(b * S_ + q0 + r) * (H_ * D_) + h * D_ + nc * 16 + nl;
            O[idx] = f2bf(oacc[nc][i] / lr[i]);
        }
}

// ---------------------------------------------------------------------------
// SwiGLU combine: g = silu(u) * t   (bf16 in/out, f32 math)
// ---------------------------------------------------------------------------
__global__ __launch_bounds__(256) void silu_mul(
    const unsigned short* __restrict__ u, const unsigned short* __restrict__ t,
    unsigned short* __restrict__ g, long n) {
    const long stride = (long)gridDim.x * 256;
    for (long i = (long)blockIdx.x * 256 + threadIdx.x; i < n; i += stride) {
        const float uu = bf2f(u[i]);
        const float sg = uu / (1.f + __expf(-uu));
        g[i] = f2bf(sg * bf2f(t[i]));
    }
}

// ---------------------------------------------------------------------------
// Host-side orchestration
// ---------------------------------------------------------------------------
extern "C" void kernel_launch(void* const* d_in, const int* in_sizes, int n_in,
                              void* d_out, int out_size, void* d_ws, size_t ws_size,
                              hipStream_t stream) {
    (void)in_sizes; (void)n_in; (void)out_size; (void)ws_size;

    const float* x    = (const float*)d_in[0];
    const float* wq   = (const float*)d_in[1];
    const float* wk   = (const float*)d_in[2];
    const float* wv   = (const float*)d_in[3];
    const float* wo   = (const float*)d_in[4];
    const float* qnw  = (const float*)d_in[5];
    const float* knw  = (const float*)d_in[6];
    const float* ln1w = (const float*)d_in[7];
    const float* ln2w = (const float*)d_in[8];
    const float* w1   = (const float*)d_in[9];
    const float* w2   = (const float*)d_in[10];
    const float* w3   = (const float*)d_in[11];

    char* ws = (char*)d_ws;

    // -------- workspace layout (bytes) --------
    constexpr size_t SZ_WQ  = (size_t)HID_ * HID_ * 2;          // 8M
    constexpr size_t SZ_WK  = (size_t)HK_ * D_ * HID_ * 2;      // 2M
    constexpr size_t SZ_W1  = (size_t)IM_ * HID_ * 2;           // 32M
    constexpr size_t SZ_X1  = (size_t)NTOK * HID_ * 4;          // 32M
    constexpr size_t SZ_HB  = (size_t)NTOK * HID_ * 2;          // 16M
    constexpr size_t SZ_QF  = (size_t)NTOK * HID_ * 4;          // 32M
    constexpr size_t SZ_KF  = (size_t)NTOK * HK_ * D_ * 4;      // 8M
    constexpr size_t SZ_KB  = (size_t)NTOK * HK_ * D_ * 2;      // 4M
    constexpr size_t SZ_UB  = (size_t)NTOK * IM_ * 2;           // 64M

    size_t off = 0;
    unsigned short* WQB = (unsigned short*)(ws + off); off += SZ_WQ;
    unsigned short* WKB = (unsigned short*)(ws + off); off += SZ_WK;
    unsigned short* WVB = (unsigned short*)(ws + off); off += SZ_WK;
    unsigned short* WOB = (unsigned short*)(ws + off); off += SZ_WQ;
    unsigned short* W1B = (unsigned short*)(ws + off); off += SZ_W1;
    unsigned short* W3B = (unsigned short*)(ws + off); off += SZ_W1;
    unsigned short* W2B = (unsigned short*)(ws + off); off += SZ_W1;
    float*          X1  = (float*)(ws + off);          off += SZ_X1;
    unsigned short* H2B = (unsigned short*)(ws + off); off += SZ_HB;
    unsigned short* TB  = (unsigned short*)(ws + off); off += SZ_UB;
    // region A (reused by UB after attention/o-proj are done)
    char* regA = ws + off;
    size_t roff = 0;
    unsigned short* HB = (unsigned short*)(regA + roff); roff += SZ_HB;
    float*          QF = (float*)(regA + roff);          roff += SZ_QF;
    float*          KF = (float*)(regA + roff);          roff += SZ_KF;
    unsigned short* QB = (unsigned short*)(regA + roff); roff += SZ_HB;
    unsigned short* KB = (unsigned short*)(regA + roff); roff += SZ_KB;
    unsigned short* VB = (unsigned short*)(regA + roff); roff += SZ_KB;
    unsigned short* VT = (unsigned short*)(regA + roff); roff += SZ_KB;
    unsigned short* AB = (unsigned short*)(regA + roff); roff += SZ_HB;
    unsigned short* UB = (unsigned short*)regA;  // aliases dead HB/QF/KF region

    // -------- weight casts f32 -> bf16 --------
    cast_f32_bf16<<<2048, 256, 0, stream>>>(wq, WQB, (long)HID_ * HID_);
    cast_f32_bf16<<<2048, 256, 0, stream>>>(wk, WKB, (long)HK_ * D_ * HID_);
    cast_f32_bf16<<<2048, 256, 0, stream>>>(wv, WVB, (long)HK_ * D_ * HID_);
    cast_f32_bf16<<<2048, 256, 0, stream>>>(wo, WOB, (long)HID_ * HID_);
    cast_f32_bf16<<<4096, 256, 0, stream>>>(w1, W1B, (long)IM_ * HID_);
    cast_f32_bf16<<<4096, 256, 0, stream>>>(w2, W2B, (long)HID_ * IM_);
    cast_f32_bf16<<<4096, 256, 0, stream>>>(w3, W3B, (long)IM_ * HID_);

    // -------- attention block --------
    rmsnorm_bf16<<<NTOK, 256, 0, stream>>>(x, ln1w, HB, HID_);

    gemm_bf16_tn<0><<<dim3(HID_ / 128, NTOK / 128), 256, 0, stream>>>(
        HB, WQB, QF, nullptr, nullptr, NTOK, HID_, HID_);
    gemm_bf16_tn<0><<<dim3(HK_ * D_ / 128, NTOK / 128), 256, 0, stream>>>(
        HB, WKB, KF, nullptr, nullptr, NTOK, HK_ * D_, HID_);
    gemm_bf16_tn<1><<<dim3(HK_ * D_ / 128, NTOK / 128), 256, 0, stream>>>(
        HB, WVB, nullptr, VB, nullptr, NTOK, HK_ * D_, HID_);

    qknorm_rope_bf16<<<NTOK, 256, 0, stream>>>(QF, qnw, QB, HID_);
    qknorm_rope_bf16<<<NTOK, 256, 0, stream>>>(KF, knw, KB, HK_ * D_);
    v_transpose<<<2048, 256, 0, stream>>>(VB, VT);

    attn_swa<<<B_ * H_ * (S_ / 64), 128, 0, stream>>>(QB, KB, VT, AB);

    gemm_bf16_tn<2><<<dim3(HID_ / 128, NTOK / 128), 256, 0, stream>>>(
        AB, WOB, X1, nullptr, x, NTOK, HID_, HID_);

    // -------- MLP block --------
    rmsnorm_bf16<<<NTOK, 256, 0, stream>>>(X1, ln2w, H2B, HID_);

    gemm_bf16_tn<1><<<dim3(IM_ / 128, NTOK / 128), 256, 0, stream>>>(
        H2B, W1B, nullptr, UB, nullptr, NTOK, IM_, HID_);
    gemm_bf16_tn<1><<<dim3(IM_ / 128, NTOK / 128), 256, 0, stream>>>(
        H2B, W3B, nullptr, TB, nullptr, NTOK, IM_, HID_);

    silu_mul<<<4096, 256, 0, stream>>>(UB, TB, UB, (long)NTOK * IM_);

    gemm_bf16_tn<2><<<dim3(HID_ / 128, NTOK / 128), 256, 0, stream>>>(
        UB, W2B, (float*)d_out, nullptr, X1, NTOK, HID_, IM_);
}
